// EdgeAwareCrossAttention_62113817034769
// MI455X (gfx1250) — matrix operand order repaired and checked by reading, not compiled, not run on hardware
//
#include <hip/hip_runtime.h>
#include <hip/hip_bf16.h>

// ---------------------------------------------------------------------------
// EdgeAwareCrossAttention, fused for MI455X (gfx1250, wave32, WMMA bf16)
// B=8 N=128 M=256 E=256 De=64 H=8 D=32 ; D==32 matches wmma 16x16x32 K step.
// Edge tile is staged LDS-resident via global_load_async_to_lds_b128 so each
// block touches its 64KB edge tile in memory exactly once.
// ---------------------------------------------------------------------------

typedef __attribute__((ext_vector_type(16))) __bf16 v16bf;
typedef __attribute__((ext_vector_type(8)))  __bf16 v8bf;
typedef __attribute__((ext_vector_type(8)))  float  v8f;
typedef __attribute__((ext_vector_type(4)))  float  v4f;

#define CB   8
#define CN   128
#define CM   256
#define CE   256
#define CDE  64
#define CH   8

// ---- WMMA helpers ---------------------------------------------------------

__device__ __forceinline__ v8f wmma_bf16(v16bf a, v16bf b, v8f c) {
  // 8 args: (neg_a, A, neg_b, B, c_mod, C, reuse_a, reuse_b)
  return __builtin_amdgcn_wmma_f32_16x16x32_bf16(false, a, false, b,
                                                 (short)0, c, false, false);
}

// Element offset inside a fragment-major B matrix.
// B is [Ktot x Ntot]; tiles are 32(K) x 16(N). Per ISA layout (16-bit B):
// lanes 0-15 hold K rows k0+{0..7,16..23} (col = lane), lanes 16-31 hold k0+{8..15,24..31}.
// Each fragment = 32 lanes x 16 bf16 contiguous (1KB).
__device__ __host__ __forceinline__ int bfrag_off(int k, int n, int ntiles) {
  int kt   = k >> 5;
  int nt   = n >> 4;
  int lane = (n & 15) + (((k >> 3) & 1) ? 16 : 0);
  int j    = (k & 7) + ((k & 16) ? 8 : 0);
  return ((kt * ntiles + nt) * 32 + lane) * 16 + j;
}

__device__ __forceinline__ v16bf loadB_frag(const __bf16* base, int fragIdx) {
  int lane = threadIdx.x & 31;
  return *(const v16bf*)(base + (fragIdx * 32 + lane) * 16);
}

// A fragment (16x32 bf16) from a row-major [16 x ld] buffer.
// ISA A layout: lane L (M = L&15): lanes<16 hold K {0..7,16..23},
// lanes>=16 hold K {8..15,24..31} -> two contiguous 16B loads per lane.
__device__ __forceinline__ v16bf loadA_rm(const __bf16* a, int ld, int k0) {
  int lane = threadIdx.x & 31;
  int mr   = lane & 15;
  int kb   = (lane < 16) ? 0 : 8;
  const __bf16* p = a + mr * ld + k0 + kb;
  union { v16bf v; v8bf h[2]; } u;
  u.h[0] = *(const v8bf*)(p);
  u.h[1] = *(const v8bf*)(p + 16);
  return u.v;
}

// ---- Kernel 1: weight prep (bf16 convert + fragment swizzle) --------------
// ws element layout (__bf16 units):
//   0       wqT_frag  (B[k=e_in, n=e_out], 65536)
//   65536   wkT_frag
//   131072  wvT_frag
//   196608  woT row-major transpose woT[e][e'] = wo_w[e'][e]   (65536)
//   262144  ep_frag   (B[k=e, n=c], 16384)
//   278528  epT_frag  (B[k=c, n=e], 16384)
//   294912  end of weight region

__global__ __launch_bounds__(256) void prep_weights(
    const float* __restrict__ wq, const float* __restrict__ wk,
    const float* __restrict__ wv, const float* __restrict__ wo,
    const float* __restrict__ epw, __bf16* __restrict__ ws0) {
  int i = blockIdx.x * blockDim.x + threadIdx.x;   // 0..294911
  if (i < 196608) {
    const float* w   = (i < 65536) ? wq : (i < 131072 ? wk : wv);
    __bf16*      dst = ws0 + (i < 65536 ? 0 : (i < 131072 ? 65536 : 131072));
    int i2 = i & 65535;
    int eo = i2 >> 8, ei = i2 & 255;               // w[eo][ei]
    dst[bfrag_off(ei, eo, 16)] = (__bf16)w[i2];
  } else if (i < 262144) {
    int i2 = i - 196608;
    int eo = i2 >> 8, ei = i2 & 255;
    ws0[196608 + ei * 256 + eo] = (__bf16)wo[i2];
  } else {
    int i2 = i - 262144;                            // 0..32767
    int which = i2 >> 14;                           // 0: ep_frag, 1: epT_frag
    int j = i2 & 16383;
    int e = j >> 6, c = j & 63;                     // epw[e][c]
    float vv = epw[j];
    if (which == 0) ws0[262144 + bfrag_off(e, c, 4)]  = (__bf16)vv;
    else            ws0[278528 + bfrag_off(c, e, 16)] = (__bf16)vv;
  }
}

// ---- Kernel 2: QKV projections via WMMA -----------------------------------
// Y[r,eo] = sum_ei X[r,ei] * W[eo,ei].  Outputs:
//  proj0: Qbf row-major bf16 [B*N, 256]
//  proj1: KhatT_frag per b: B[k=e(256), n=m(256)] fragment-major
//  proj2: Vhat_frag  per b: B[k=m(256), n=e(256)] fragment-major

__global__ __launch_bounds__(256) void project_qkv(
    const float* __restrict__ q, const float* __restrict__ k,
    const float* __restrict__ v, const __bf16* __restrict__ ws0,
    __bf16* __restrict__ Qbf, __bf16* __restrict__ KTf,
    __bf16* __restrict__ Vf) {
  int proj = blockIdx.y;                 // 0=q,1=k,2=v
  int rt   = blockIdx.x;                 // 16-row tile
  int rowTiles = (proj == 0) ? (CB * CN / 16) : (CB * CM / 16);
  if (rt >= rowTiles) return;
  const float*  X  = (proj == 0) ? q : (proj == 1 ? k : v);
  const __bf16* WT = ws0 + proj * 65536;

  __shared__ __align__(32) __bf16 Atile[16 * 256];
  int tid = threadIdx.x, wave = tid >> 5, lane = tid & 31;

  for (int i = tid; i < 4096; i += 256)
    Atile[i] = (__bf16)X[(size_t)(rt * 16 + (i >> 8)) * 256 + (i & 255)];
  __syncthreads();

  for (int t = 0; t < 2; ++t) {
    int et = wave + t * 8;               // output e-tile 0..15
    v8f acc = {};
    v16bf bcur = loadB_frag(WT, et);     // double-buffered B stream
    for (int kk = 0; kk < 8; ++kk) {
      v16bf bnext = (kk < 7) ? loadB_frag(WT, (kk + 1) * 16 + et) : bcur;
      v16bf a = loadA_rm(Atile, 256, kk * 32);
      acc = wmma_bf16(a, bcur, acc);
      bcur = bnext;
    }
    int col = et * 16 + (lane & 15);
    for (int g = 0; g < 8; ++g) {
      int rowl = g + ((lane < 16) ? 0 : 8);    // C layout: VGPR g -> M=g / g+8
      int r    = rt * 16 + rowl;
      __bf16 val = (__bf16)acc[g];
      if (proj == 0) {
        Qbf[(size_t)r * 256 + col] = val;
      } else {
        int b_ = r >> 8, m = r & 255;
        if (proj == 1) KTf[(size_t)b_ * 65536 + bfrag_off(col, m, 16)] = val;
        else           Vf [(size_t)b_ * 65536 + bfrag_off(m, col, 16)] = val;
      }
    }
  }
}

// ---- Kernel 3: fused edge-aware attention, one block per (b,n) ------------
// Dynamic LDS layout (bytes):
//   0      stage   fp32 edge tile [256 x 64]            65536
//   65536  edgeB   bf16 fragment-major edge (reused)    32768
//   98304  Aq      bf16 masked-Q A matrix [16 x 256]     8192
//   106496 attnA   bf16 attention A matrix [16 x 256]    8192
//   114688 qepA    bf16 [16 x 64]                        2048
//   116736 aeA     bf16 [16 x 64]                        2048
//   118784 scores  f32 [8 x 256]                         8192
//   126976 ctx     f32 [256]                             1024
//   128000 red     f32 [2]                                  8
#define SMEM_BYTES 131072

__global__ __launch_bounds__(256) void attn_main(
    const float* __restrict__ qin, const float* __restrict__ edge,
    const __bf16* __restrict__ ws0, const __bf16* __restrict__ Qbf,
    const __bf16* __restrict__ KTf, const __bf16* __restrict__ Vf,
    const float* __restrict__ epb, const float* __restrict__ wob,
    const float* __restrict__ lng, const float* __restrict__ lnb,
    float* __restrict__ out) {
  extern __shared__ __align__(128) char smem[];
  float*  stage   = (float*)(smem);
  __bf16* edgeB   = (__bf16*)(smem + 65536);
  __bf16* Aq      = (__bf16*)(smem + 98304);
  __bf16* attnA   = (__bf16*)(smem + 106496);
  __bf16* qepA    = (__bf16*)(smem + 114688);
  __bf16* aeA     = (__bf16*)(smem + 116736);
  float*  scoresS = (float*)(smem + 118784);
  float*  ctxS    = (float*)(smem + 126976);
  float*  red     = (float*)(smem + 128000);

  const __bf16* epF = ws0 + 262144;
  const __bf16* epT = ws0 + 278528;
  const __bf16* woT = ws0 + 196608;

  int bn = blockIdx.x;                   // 0..1023
  int b  = bn >> 7;
  int tid = threadIdx.x, wave = tid >> 5, lane = tid & 31;

  const float*  eptr = edge + (size_t)bn * (CM * CDE);
  const __bf16* Qrow = Qbf + (size_t)bn * 256;
  const __bf16* KT_b = KTf + (size_t)b * 65536;
  const __bf16* V_b  = Vf  + (size_t)b * 65536;

  // Phase B0: async-stage the raw fp32 edge tile into LDS (ASYNCcnt path).
  // Each thread issues 16 x 16B async copies; it later converts exactly the
  // chunks its own lanes loaded, so a per-wave s_wait_asynccnt suffices.
  for (int it = 0; it < 16; ++it) {
    int i4 = tid + it * 256;
    unsigned long long g = (unsigned long long)(uintptr_t)(eptr + (size_t)i4 * 4);
    unsigned int       l = (unsigned int)(uintptr_t)(stage + (size_t)i4 * 4);
    asm volatile("global_load_async_to_lds_b128 %0, %1, off"
                 :: "v"(l), "v"(g) : "memory");
  }

  // Phase A (overlaps async): masked-Q A matrix, zero attnA, zero reductions.
  for (int i = tid; i < 4096; i += 256) {
    int row = i >> 8, e = i & 255;
    __bf16 z = (__bf16)0.0f;
    Aq[i]    = (row < 8 && (e >> 5) == row) ? Qrow[e] : z;
    attnA[i] = z;
  }
  if (tid < 2) red[tid] = 0.0f;

  asm volatile("s_wait_asynccnt 0x0" ::: "memory");

  // Phase B1: build edge orientation for scores: B[k=c(64), n=m(256)].
  for (int it = 0; it < 16; ++it) {
    int i4 = tid + it * 256;
    v4f d = *(const v4f*)(stage + (size_t)i4 * 4);
    int f0 = i4 * 4;
    for (int j = 0; j < 4; ++j) {
      int f = f0 + j, m = f >> 6, c = f & 63;
      edgeB[bfrag_off(c, m, 16)] = (__bf16)d[j];
    }
  }
  __syncthreads();

  // Phase C (waves 0-3): qep[16,64] = Aq[16,256] @ ep_frag[256,64]
  if (wave < 4) {
    v8f acc = {};
    for (int kk = 0; kk < 8; ++kk) {
      v16bf a = loadA_rm(Aq, 256, kk * 32);
      v16bf bf = loadB_frag(epF, kk * 4 + wave);
      acc = wmma_bf16(a, bf, acc);
    }
    int col = wave * 16 + (lane & 15);
    for (int g = 0; g < 8; ++g) {
      int row = g + ((lane < 16) ? 0 : 8);
      qepA[row * 64 + col] = (__bf16)acc[g];     // rows 8-15 are zeros
    }
  }
  __syncthreads();

  // Phase D: scores[h,m] = (Aq@K^T + qep@edge^T) / sqrt(D)
  const float invScale = 0.17677669529663687f;   // 1/sqrt(32)
  for (int t = 0; t < 2; ++t) {
    int mt = wave + t * 8;
    v8f acc = {};
    v16bf bcur = loadB_frag(KT_b, mt);
    for (int kk = 0; kk < 8; ++kk) {
      if (kk < 7) __builtin_prefetch(KT_b + ((kk + 1) * 16 + mt) * 512, 0, 3);
      v16bf bnext = (kk < 7) ? loadB_frag(KT_b, (kk + 1) * 16 + mt) : bcur;
      v16bf a = loadA_rm(Aq, 256, kk * 32);
      acc = wmma_bf16(a, bcur, acc);
      bcur = bnext;
    }
    for (int k2 = 0; k2 < 2; ++k2) {
      v16bf a = loadA_rm(qepA, 64, k2 * 32);
      v16bf bf = loadB_frag(edgeB, k2 * 16 + mt);   // ds_load_b128
      acc = wmma_bf16(a, bf, acc);
    }
    if (lane < 16)                                  // rows 0..7 = heads
      for (int g = 0; g < 8; ++g)
        scoresS[g * 256 + mt * 16 + lane] = acc[g] * invScale;
  }
  __syncthreads();

  // Phase B2: rebuild edge tile from the LDS fp32 stage (no global re-read),
  // orientation for ae: B[k=m(256), n=c(64)].
  for (int it = 0; it < 16; ++it) {
    int i4 = tid + it * 256;
    v4f d = *(const v4f*)(stage + (size_t)i4 * 4);
    int f0 = i4 * 4;
    for (int j = 0; j < 4; ++j) {
      int f = f0 + j, m = f >> 6, c = f & 63;
      edgeB[bfrag_off(m, c, 4)] = (__bf16)d[j];
    }
  }

  // Phase E: softmax over m, wave w <-> head w (wave32 shuffle reductions).
  {
    int h = wave;
    float vals[8];
    float mx = -3.0e38f;
    for (int i = 0; i < 8; ++i) {
      vals[i] = scoresS[h * 256 + lane + i * 32];
      mx = fmaxf(mx, vals[i]);
    }
    for (int s = 16; s > 0; s >>= 1) mx = fmaxf(mx, __shfl_xor(mx, s, 32));
    float sum = 0.0f;
    for (int i = 0; i < 8; ++i) { vals[i] = __expf(vals[i] - mx); sum += vals[i]; }
    for (int s = 16; s > 0; s >>= 1) sum += __shfl_xor(sum, s, 32);
    float inv = 1.0f / sum;
    for (int i = 0; i < 8; ++i)
      attnA[h * 256 + lane + i * 32] = (__bf16)(vals[i] * inv);
  }
  __syncthreads();

  // Phase F: ctx (AV part) + ae = attn @ edge
  for (int t = 0; t < 2; ++t) {
    int et = wave + t * 8;
    v8f acc = {};
    v16bf bcur = loadB_frag(V_b, et);
    for (int kk = 0; kk < 8; ++kk) {
      if (kk < 7) __builtin_prefetch(V_b + ((kk + 1) * 16 + et) * 512, 0, 3);
      v16bf bnext = (kk < 7) ? loadB_frag(V_b, (kk + 1) * 16 + et) : bcur;
      v16bf a = loadA_rm(attnA, 256, kk * 32);
      acc = wmma_bf16(a, bcur, acc);
      bcur = bnext;
    }
    int h = et >> 1;                      // head owning this e-tile
    if (lane < 16) ctxS[et * 16 + lane] = acc[h];
  }
  if (wave < 4) {
    v8f acc = {};
    for (int kk = 0; kk < 8; ++kk) {
      v16bf a = loadA_rm(attnA, 256, kk * 32);
      v16bf bf = loadB_frag(edgeB, kk * 4 + wave);   // ds_load_b128
      acc = wmma_bf16(a, bf, acc);
    }
    int col = wave * 16 + (lane & 15);
    for (int g = 0; g < 8; ++g) {
      int row = g + ((lane < 16) ? 0 : 8);
      aeA[row * 64 + col] = (__bf16)acc[g];
    }
  }
  __syncthreads();

  // Phase G: ctx += ae @ ep_w (epT_frag) + ep_b   (Sum attn = 1)
  for (int t = 0; t < 2; ++t) {
    int et = wave + t * 8;
    v8f acc = {};
    for (int k2 = 0; k2 < 2; ++k2) {
      v16bf a = loadA_rm(aeA, 64, k2 * 32);
      v16bf bf = loadB_frag(epT, k2 * 16 + et);
      acc = wmma_bf16(a, bf, acc);
    }
    int h = et >> 1;
    if (lane < 16) {
      int e = et * 16 + lane;
      ctxS[e] += acc[h] + epb[e];
    }
  }
  __syncthreads();

  // Phase H: out = LN(q + ctx @ wo^T + wo_b)
  float o = wob[tid];
  for (int e = 0; e < 256; ++e)
    o += ctxS[e] * (float)woT[e * 256 + tid];
  float r = qin[(size_t)bn * 256 + tid] + o;
  atomicAdd(&red[0], r);                 // ds_add_f32
  atomicAdd(&red[1], r * r);
  __syncthreads();
  float mu  = red[0] * (1.0f / 256.0f);
  float var = red[1] * (1.0f / 256.0f) - mu * mu;
  float ov  = rsqrtf(var + 1e-5f);
  out[(size_t)bn * 256 + tid] = (r - mu) * ov * lng[tid] + lnb[tid];
}

// ---------------------------------------------------------------------------

extern "C" void kernel_launch(void* const* d_in, const int* in_sizes, int n_in,
                              void* d_out, int out_size, void* d_ws, size_t ws_size,
                              hipStream_t stream) {
  const float* q    = (const float*)d_in[0];
  const float* k    = (const float*)d_in[1];
  const float* v    = (const float*)d_in[2];
  const float* edge = (const float*)d_in[3];
  const float* wq   = (const float*)d_in[4];
  const float* wk   = (const float*)d_in[5];
  const float* wv   = (const float*)d_in[6];
  const float* wo_w = (const float*)d_in[7];
  const float* wo_b = (const float*)d_in[8];
  const float* ep_w = (const float*)d_in[9];
  const float* ep_b = (const float*)d_in[10];
  const float* ln_g = (const float*)d_in[11];
  const float* ln_b = (const float*)d_in[12];

  __bf16* ws0 = (__bf16*)d_ws;           // weights: 294912 elems
  __bf16* Qbf = ws0 + 294912;            // 1024*256
  __bf16* KTf = Qbf + 262144;            // 8 * 65536
  __bf16* Vf  = KTf + 524288;            // 8 * 65536   (~3.2 MB total)

  prep_weights<<<1152, 256, 0, stream>>>(wq, wk, wv, wo_w, ep_w, ws0);
  project_qkv<<<dim3(128, 3), 256, 0, stream>>>(q, k, v, ws0, Qbf, KTf, Vf);
  attn_main<<<CB * CN, 256, SMEM_BYTES, stream>>>(q, edge, ws0, Qbf, KTf, Vf,
                                                  ep_b, wo_b, ln_g, ln_b,
                                                  (float*)d_out);
}
